// ForgetfulCausalTopKAttention_30270929502398
// MI455X (gfx1250) — matrix operand order repaired and checked by reading, not compiled, not run on hardware
//
#include <hip/hip_runtime.h>

typedef __attribute__((ext_vector_type(16))) __bf16   v16bf;
typedef __attribute__((ext_vector_type(8)))  float    v8f;
typedef __attribute__((ext_vector_type(4)))  unsigned v4u;

namespace {

constexpr int kBatch = 2;
constexpr int kSeq   = 2048;
constexpr int kEmbed = 2048;
constexpr int kLat   = 2048;
constexpr int kHeads = 16;
constexpr int kHd    = 128;
constexpr int kPers  = 4;
constexpr int kTopK  = 64;
constexpr int kM     = kBatch * kSeq;   // 4096
constexpr int kSkp   = 2080;            // keys 2052 padded to /32
constexpr float kScale = 0.08838834764831845f; // 1/sqrt(128)

static_assert(sizeof(v16bf) == 32, "v16bf must be 32B (8 VGPRs)");

union FragU {
  struct { v4u lo, hi; } p;
  v16bf v;
};

__device__ __forceinline__ v8f vzero8() {
  v8f z = {0.f, 0.f, 0.f, 0.f, 0.f, 0.f, 0.f, 0.f};
  return z;
}

__device__ __forceinline__ unsigned short f2bf(float f) {
  unsigned u = __float_as_uint(f);
  u += 0x7FFFu + ((u >> 16) & 1u);          // round-to-nearest-even
  return (unsigned short)(u >> 16);
}

// monotone mapping float -> u32 (larger float => larger key)
__device__ __forceinline__ unsigned mono_key(float f) {
  unsigned u = __float_as_uint(f);
  return (u & 0x80000000u) ? ~u : (u | 0x80000000u);
}

__device__ __forceinline__ v8f wmma_bf16(const FragU& a, const FragU& b, v8f c) {
  return __builtin_amdgcn_wmma_f32_16x16x32_bf16(false, a.v, false, b.v,
                                                 (short)0, c, false, false);
}

__device__ __forceinline__ void lds_fence() {
  asm volatile("s_wait_dscnt 0x0" ::: "memory");
}

// ---------------- pack / convert kernels ----------------

__global__ void convert_bf16_kernel(const float* __restrict__ in,
                                    unsigned short* __restrict__ out, int n) {
  int i = blockIdx.x * blockDim.x + threadIdx.x;
  if (i < n) out[i] = f2bf(in[i]);
}

// in: [2048][2048] row major; out[n][k] = in[k][n]
__global__ void transpose_bf16_kernel(const float* __restrict__ in,
                                      unsigned short* __restrict__ out) {
  int i = blockIdx.x * blockDim.x + threadIdx.x;
  if (i < 2048 * 2048) {
    int n = i >> 11, k = i & 2047;
    out[i] = f2bf(in[((size_t)k << 11) + n]);
  }
}

// Append persistent tokens (rows/cols 2048..2051) and zero padding up to 2080.
__global__ void fill_persistent_kernel(const float* __restrict__ pk,
                                       const float* __restrict__ pv,
                                       unsigned short* __restrict__ Kb,  // [BH][kSkp][128]
                                       unsigned short* __restrict__ Vt)  // [BH][128][kSkp]
{
  int idx = blockIdx.x * blockDim.x + threadIdx.x;
  if (idx >= kBatch * kHeads * 32 * kHd) return;
  int d  = idx & 127;
  int i  = (idx >> 7) & 31;       // pad slot -> key row 2048+i
  int bh = idx >> 12;
  int h  = bh & 15;
  unsigned short kv = 0, vv = 0;
  if (i < kPers) {
    kv = f2bf(pk[(size_t)i * kLat + h * kHd + d]);
    vv = f2bf(pv[(size_t)i * kLat + h * kHd + d]);
  }
  Kb[((size_t)bh * kSkp + kSeq + i) * kHd + d] = kv;
  Vt[((size_t)bh * kHd + d) * kSkp + kSeq + i] = vv;
}

// ---------------- WMMA GEMM ----------------
// C[4096][2048] = A[4096][2048](bf16) * Bw^T  (Bw stored pre-transposed [N][K], bf16) + bias
// Wave tile: 32(M) x 64(N)  ->  8 WMMAs per 32-deep k-step, AI ~341 FLOP/B from L2.
// MODE 0: Q  -> bf16 [bh][2048][128]
// MODE 1: K  -> bf16 [bh][2080][128]
// MODE 2: Vt -> bf16 [bh][128][2080]   (transposed store)
// MODE 3: f32 row-major [M][N]
template <int MODE>
__global__ __launch_bounds__(128)
void gemm_kernel(const unsigned short* __restrict__ A,
                 const unsigned short* __restrict__ Bw,
                 const float* __restrict__ bias,
                 void* __restrict__ Out)
{
  constexpr int Kd = 2048, N = 2048;
  const int lane = threadIdx.x & 31;
  const int wid  = threadIdx.x >> 5;
  const int lr   = lane & 15;
  const int hiA  = (lane >> 4) * 8;    // A k-chunk shift: {0..7,16..23} vs {8..15,24..31}
  const int hiB  = (lane >> 4) * 8;    // B k-chunk shift: {0..7,16..23} vs {8..15,24..31}
  const int tm = blockIdx.y * 32;
  const int tn = blockIdx.x * 256 + wid * 64;

  const unsigned short* arow0 = A + (size_t)(tm + lr) * Kd;
  const unsigned short* arow1 = A + (size_t)(tm + 16 + lr) * Kd;
  v8f acc[2][4];
#pragma unroll
  for (int mi = 0; mi < 2; ++mi)
#pragma unroll
    for (int t = 0; t < 4; ++t) acc[mi][t] = vzero8();

  for (int kk = 0; kk < Kd; kk += 32) {
    FragU a0, a1;
    a0.p.lo = *(const v4u*)(arow0 + kk + hiA);
    a0.p.hi = *(const v4u*)(arow0 + kk + 16 + hiA);
    a1.p.lo = *(const v4u*)(arow1 + kk + hiA);
    a1.p.hi = *(const v4u*)(arow1 + kk + 16 + hiA);
#pragma unroll
    for (int t = 0; t < 4; ++t) {
      const unsigned short* brow = Bw + (size_t)(tn + 16 * t + lr) * Kd + kk + hiB;
      FragU bf;
      bf.p.lo = *(const v4u*)(brow);
      bf.p.hi = *(const v4u*)(brow + 16);
      acc[0][t] = wmma_bf16(a0, bf, acc[0][t]);
      acc[1][t] = wmma_bf16(a1, bf, acc[1][t]);
    }
  }

#pragma unroll
  for (int mi = 0; mi < 2; ++mi) {
    const int mbase = tm + mi * 16 + (lane >> 4) * 8; // D: vgpr j -> row mbase+j, col = lane%16
#pragma unroll
    for (int t = 0; t < 4; ++t) {
      const int n = tn + 16 * t + lr;
      const float bs = bias[n];
      if constexpr (MODE == 3) {
        float* o = (float*)Out;
#pragma unroll
        for (int j = 0; j < 8; ++j)
          o[(size_t)(mbase + j) * N + n] = acc[mi][t][j] + bs;
      } else if constexpr (MODE == 2) {
        unsigned short* o = (unsigned short*)Out;
        const int bb = mbase >> 11, s0 = mbase & 2047;
        const int h = n >> 7, d = n & 127;
        v4u packed;
#pragma unroll
        for (int w = 0; w < 4; ++w) {
          unsigned loh = (unsigned)f2bf(acc[mi][t][2 * w] + bs);
          unsigned hih = (unsigned)f2bf(acc[mi][t][2 * w + 1] + bs);
          packed[w] = loh | (hih << 16);
        }
        *(v4u*)(o + ((size_t)((bb * kHeads + h) * kHd + d)) * kSkp + s0) = packed;
      } else {
        unsigned short* o = (unsigned short*)Out;
        constexpr int ROWS = (MODE == 0) ? kSeq : kSkp;
        const int h = n >> 7, d = n & 127;
#pragma unroll
        for (int j = 0; j < 8; ++j) {
          const int m = mbase + j;
          const int bb = m >> 11, s = m & 2047;
          o[((size_t)((bb * kHeads + h) * ROWS + s)) * kHd + d] = f2bf(acc[mi][t][j] + bs);
        }
      }
    }
  }
}

// ---------------- attention (scores + top-k softmax + attn@V) ----------------
// block = 256 threads (8 waves), one block per (b, h, 16-query tile)
__global__ __launch_bounds__(256)
void attn_kernel(const unsigned short* __restrict__ Q,   // [BH][2048][128]
                 const unsigned short* __restrict__ Kb,  // [BH][2080][128]
                 const unsigned short* __restrict__ Vt,  // [BH][128][2080]
                 unsigned short* __restrict__ O)         // [B*S][2048] bf16
{
  extern __shared__ char smem[];
  float*          sc   = (float*)smem;                                  // [16][kSkp]
  unsigned short* pr   = (unsigned short*)(smem + 16 * kSkp * 4);       // [16][kSkp]
  unsigned*       hist = (unsigned*)(smem + 16 * kSkp * 4 + 16 * kSkp * 2); // [8][256]

  const int blk = blockIdx.x;
  const int qt  = blk & 127;         // query tile within (b,h)
  const int bh  = blk >> 7;
  const int b   = bh >> 4, h = bh & 15;
  const int lane = threadIdx.x & 31;
  const int wid  = threadIdx.x >> 5;
  const int lr   = lane & 15;
  const int hiA  = (lane >> 4) * 8;
  const int hiB  = (lane >> 4) * 8;

  const int Lv  = 16 * (qt + 2);                 // score columns computed (covers max col 16qt+19)
  const int KT  = (16 * qt + 19) / 32 + 1;       // 32-wide key tiles for phase 2
  const int L32 = 32 * KT;

  // --- preload Q A-fragments (4 k-steps over head_dim 128) ---
  const unsigned short* qrow = Q + ((size_t)bh * kSeq + 16 * qt + lr) * kHd;
  FragU qf[4];
#pragma unroll
  for (int ks = 0; ks < 4; ++ks) {
    qf[ks].p.lo = *(const v4u*)(qrow + 32 * ks + hiA);
    qf[ks].p.hi = *(const v4u*)(qrow + 32 * ks + 16 + hiA);
  }

  // --- phase 1: scores -> LDS (masked, scaled) ---
  const unsigned short* kbase = Kb + (size_t)bh * kSkp * kHd;
  for (int jt = wid; jt < qt + 2; jt += 8) {
    v8f acc = vzero8();
    const unsigned short* krow = kbase + (size_t)(16 * jt + lr) * kHd;
#pragma unroll
    for (int ks = 0; ks < 4; ++ks) {
      FragU bf;
      bf.p.lo = *(const v4u*)(krow + 32 * ks + hiB);
      bf.p.hi = *(const v4u*)(krow + 32 * ks + hiB + 16);
      acc = wmma_bf16(qf[ks], bf, acc);
    }
    const int c = 16 * jt + lr;
    const int m0 = (lane >> 4) * 8;
#pragma unroll
    for (int j = 0; j < 8; ++j) {
      const int m = m0 + j;
      const int r = 16 * qt + m;
      float s = acc[j] * kScale;
      if (c >= kPers && (c - kPers) > r) s = -__builtin_inff();  // pad cols auto-masked
      sc[(size_t)m * kSkp + c] = s;
    }
  }
  __syncthreads();

  // --- phase 2: per-row top-K threshold (radix select) + softmax -> bf16 probs ---
  for (int rr = 0; rr < 2; ++rr) {
    const int row = wid + rr * 8;
    float* srow = sc + (size_t)row * kSkp;
    unsigned thresh = 0u;                       // keep everything if Lv <= K
    if (Lv > kTopK) {
      unsigned* hh = hist + wid * 256;
      unsigned prefix = 0;
      int krem = kTopK;
      for (int pass = 3; pass >= 0; --pass) {
        const int shift = pass * 8;
        for (int i = lane; i < 256; i += 32) hh[i] = 0u;
        lds_fence();
        for (int i = lane; i < Lv; i += 32) {
          unsigned key = mono_key(srow[i]);
          bool match = (pass == 3) || ((key >> (shift + 8)) == prefix);
          if (match) atomicAdd(&hh[(key >> shift) & 255u], 1u);
        }
        lds_fence();
        unsigned cum = 0, chosen = 0;           // all lanes scan identically
        for (int bq = 255; bq >= 0; --bq) {
          unsigned cb = hh[bq];
          if (cum + cb >= (unsigned)krem) { chosen = (unsigned)bq; break; }
          cum += cb;
        }
        krem -= (int)cum;
        prefix = (prefix << 8) | chosen;
      }
      thresh = prefix;                          // key of K-th largest
    }
    // row max (top-1 is always kept, so max over all == max over kept)
    float mx = -__builtin_inff();
    for (int i = lane; i < Lv; i += 32) mx = fmaxf(mx, srow[i]);
#pragma unroll
    for (int off = 16; off > 0; off >>= 1) mx = fmaxf(mx, __shfl_xor(mx, off, 32));
    // exp over kept entries, sum
    float sum = 0.f;
    for (int i = lane; i < Lv; i += 32) {
      float s = srow[i];
      float e = (mono_key(s) >= thresh) ? exp2f((s - mx) * 1.44269504f) : 0.f;
      srow[i] = e;
      sum += e;
    }
#pragma unroll
    for (int off = 16; off > 0; off >>= 1) sum += __shfl_xor(sum, off, 32);
    const float inv = 1.f / sum;
    unsigned short* prow = pr + (size_t)row * kSkp;
    for (int i = lane; i < L32; i += 32) {
      float e = (i < Lv) ? srow[i] * inv : 0.f;
      prow[i] = f2bf(e);
    }
  }
  __syncthreads();

  // --- phase 3: out_tile[16 x 16] = P[16 x L32] @ V[L32 x 16], one d-tile per wave ---
  {
    const int dt = wid;                                  // 8 waves x 16 cols = 128 head dims
    const unsigned short* vrow = Vt + ((size_t)bh * kHd + 16 * dt + lr) * kSkp;
    v8f acc = vzero8();
    for (int kt = 0; kt < KT; ++kt) {
      FragU af, bf;
      const unsigned short* prow = pr + (size_t)lr * kSkp + 32 * kt;
      af.p.lo = *(const v4u*)(prow + hiA);               // LDS -> ds_load_b128
      af.p.hi = *(const v4u*)(prow + 16 + hiA);
      bf.p.lo = *(const v4u*)(vrow + 32 * kt + hiB);
      bf.p.hi = *(const v4u*)(vrow + 32 * kt + hiB + 16);
      acc = wmma_bf16(af, bf, acc);
    }
    const int col = h * kHd + 16 * dt + lr;
    const int m0  = 16 * qt + (lane >> 4) * 8;
#pragma unroll
    for (int j = 0; j < 8; ++j)
      O[((size_t)(b * kSeq + m0 + j)) * kLat + col] = f2bf(acc[j]);
  }
}

constexpr size_t kAttnSmem = (size_t)16 * kSkp * 4 + (size_t)16 * kSkp * 2 + 8 * 256 * 4; // 207,872 B

} // namespace

extern "C" void kernel_launch(void* const* d_in, const int* in_sizes, int n_in,
                              void* d_out, int out_size, void* d_ws, size_t ws_size,
                              hipStream_t stream)
{
  (void)in_sizes; (void)n_in; (void)out_size;

  const float* x  = (const float*)d_in[0];
  const float* Wq = (const float*)d_in[1];
  const float* bq = (const float*)d_in[2];
  const float* Wk = (const float*)d_in[3];
  const float* bk = (const float*)d_in[4];
  const float* Wv = (const float*)d_in[5];
  const float* bv = (const float*)d_in[6];
  const float* Wo = (const float*)d_in[7];
  const float* bo = (const float*)d_in[8];
  const float* pk = (const float*)d_in[9];
  const float* pv = (const float*)d_in[10];
  float* out = (float*)d_out;

  // workspace carve (all bf16 scratch; ~113 MB total)
  char* w = (char*)d_ws;
  auto carve = [&](size_t bytes) -> char* {
    char* p = w;
    w += (bytes + 255) & ~(size_t)255;
    return p;
  };
  unsigned short* xb  = (unsigned short*)carve((size_t)kM * kEmbed * 2);
  unsigned short* wqt = (unsigned short*)carve((size_t)kEmbed * kLat * 2);
  unsigned short* wkt = (unsigned short*)carve((size_t)kEmbed * kLat * 2);
  unsigned short* wvt = (unsigned short*)carve((size_t)kEmbed * kLat * 2);
  unsigned short* wot = (unsigned short*)carve((size_t)kLat * kEmbed * 2);
  unsigned short* Qb  = (unsigned short*)carve((size_t)kBatch * kHeads * kSeq * kHd * 2);
  unsigned short* Kb  = (unsigned short*)carve((size_t)kBatch * kHeads * kSkp * kHd * 2);
  unsigned short* Vtb = (unsigned short*)carve((size_t)kBatch * kHeads * kHd * kSkp * 2);
  unsigned short* Ob  = (unsigned short*)carve((size_t)kM * kLat * 2);
  if ((size_t)(w - (char*)d_ws) > ws_size) return;  // workspace too small: bail deterministically

  const int nx = kM * kEmbed;
  convert_bf16_kernel<<<(nx + 255) / 256, 256, 0, stream>>>(x, xb, nx);
  const int nw = 2048 * 2048;
  transpose_bf16_kernel<<<(nw + 255) / 256, 256, 0, stream>>>(Wq, wqt);
  transpose_bf16_kernel<<<(nw + 255) / 256, 256, 0, stream>>>(Wk, wkt);
  transpose_bf16_kernel<<<(nw + 255) / 256, 256, 0, stream>>>(Wv, wvt);
  transpose_bf16_kernel<<<(nw + 255) / 256, 256, 0, stream>>>(Wo, wot);

  dim3 ggrid(2048 / 256, kM / 32);   // (8, 128): 32x256 block tile, 32x64 per wave
  gemm_kernel<0><<<ggrid, 128, 0, stream>>>(xb, wqt, bq, (void*)Qb);
  gemm_kernel<1><<<ggrid, 128, 0, stream>>>(xb, wkt, bk, (void*)Kb);
  gemm_kernel<2><<<ggrid, 128, 0, stream>>>(xb, wvt, bv, (void*)Vtb);

  const int np = kBatch * kHeads * 32 * kHd;
  fill_persistent_kernel<<<(np + 255) / 256, 256, 0, stream>>>(pk, pv, Kb, Vtb);

  attn_kernel<<<kBatch * kHeads * (kSeq / 16), 256, kAttnSmem, stream>>>(Qb, Kb, Vtb, Ob);

  gemm_kernel<3><<<ggrid, 128, 0, stream>>>(Ob, wot, bo, (void*)out);
}
